// RecordingStableAudioIPAttnProcessor2_0_13211319402791
// MI455X (gfx1250) — matrix operand order repaired and checked by reading, not compiled, not run on hardware
//
#include <hip/hip_runtime.h>

// ---------------- types / helpers ----------------
typedef __attribute__((ext_vector_type(16))) __bf16       v16bf;
typedef __attribute__((ext_vector_type(8)))  float        v8f;
typedef __attribute__((ext_vector_type(8)))  unsigned int v8u;

static __device__ __forceinline__ unsigned short f2b(float f) {
    unsigned int u = __float_as_uint(f);
    unsigned int r = u + 0x7FFFu + ((u >> 16) & 1u); // round-to-nearest-even
    return (unsigned short)(r >> 16);
}

// Build a 16xbf16 WMMA fragment from two contiguous 16B chunks.
static __device__ __forceinline__ v16bf frag_from(const unsigned short* p0,
                                                  const unsigned short* p1) {
    uint4 x = *(const uint4*)p0;
    uint4 y = *(const uint4*)p1;
    v8u u = {x.x, x.y, x.z, x.w, y.x, y.y, y.z, y.w};
    return __builtin_bit_cast(v16bf, u);
}

static __device__ __forceinline__ v8f wmma_bf16(v16bf a, v16bf b, v8f c) {
    return __builtin_amdgcn_wmma_f32_16x16x32_bf16(false, a, false, b,
                                                   (short)0, c, false, false);
}

static __device__ __forceinline__ float redmax16(float v) {
    v = fmaxf(v, __shfl_xor(v, 1, 32));
    v = fmaxf(v, __shfl_xor(v, 2, 32));
    v = fmaxf(v, __shfl_xor(v, 4, 32));
    v = fmaxf(v, __shfl_xor(v, 8, 32));
    return v;
}
static __device__ __forceinline__ float redsum16(float v) {
    v += __shfl_xor(v, 1, 32);
    v += __shfl_xor(v, 2, 32);
    v += __shfl_xor(v, 4, 32);
    v += __shfl_xor(v, 8, 32);
    return v;
}

// ---------------- problem constants ----------------
#define BATCH  4
#define SQ     2048
#define DMODEL 1536
#define HEADS  24
#define KVH    12
#define HD     64
#define CROSS  768
#define TEXTL  512
#define IPL    64
#define ENCL   576
#define SCALE  0.125f   // 1/sqrt(64)

// ---------------- conversion kernels ----------------
__global__ void cvt_bf16_kernel(const float* __restrict__ in,
                                unsigned short* __restrict__ out, int n) {
    int i = blockIdx.x * blockDim.x + threadIdx.x;
    if (i < n) out[i] = f2b(in[i]);
}

// encoder [B,576,768] -> text bf16 [B*512,768], ip bf16 [B*64,768]
__global__ void enc_split_kernel(const float* __restrict__ enc,
                                 unsigned short* __restrict__ text,
                                 unsigned short* __restrict__ ip) {
    int i = blockIdx.x * blockDim.x + threadIdx.x;
    int total = BATCH * ENCL * CROSS;
    if (i >= total) return;
    int col = i % CROSS;
    int row = i / CROSS;
    int b = row / ENCL, r = row % ENCL;
    unsigned short v = f2b(enc[i]);
    if (r < TEXTL) text[((size_t)(b * TEXTL + r)) * CROSS + col] = v;
    else           ip[((size_t)(b * IPL + (r - TEXTL))) * CROSS + col] = v;
}

// RoPE on first 32 dims of each head, fold score scale, pack to bf16.
__global__ void rope_pack_kernel(const float* __restrict__ qf,
                                 const float* __restrict__ rc,
                                 const float* __restrict__ rs,
                                 unsigned short* __restrict__ qb) {
    int i = blockIdx.x * blockDim.x + threadIdx.x;
    int total = BATCH * SQ * DMODEL;
    if (i >= total) return;
    int col = i % DMODEL;
    int row = i / DMODEL;
    int d = col & (HD - 1);
    int qpos = row & (SQ - 1);
    float x = qf[i];
    float val;
    if (d < 32) {
        float c = rc[qpos * 32 + d];
        float s = rs[qpos * 32 + d];
        float rot = (d < 16) ? -qf[i + 16] : qf[i - 16];
        val = x * c + rot * s;
    } else {
        val = x;
    }
    qb[i] = f2b(val * SCALE);
}

// ---------------- tiled bf16 WMMA GEMM ----------------
// C[M,N] = A[M,K] * B[K,N].  mode 0: f32 out; 1: bf16 out; 2: f32 + bias + residual.
// 256 threads = 8 waves; block tile 128x128; per-wave 32x64 (8 C tiles); BK=32.
__global__ void __launch_bounds__(256)
gemm_bf16_wmma(const unsigned short* __restrict__ A,
               const unsigned short* __restrict__ Bw,
               float* __restrict__ Cf, unsigned short* __restrict__ Cb,
               const float* __restrict__ bias, const float* __restrict__ resid,
               int M, int N, int K, int mode) {
    __shared__ unsigned short As[128 * 32];  // [m][k]
    __shared__ unsigned short Bs[128 * 32];  // [n][k]  (transposed for contiguous frags)

    const int n0 = blockIdx.x * 128;
    const int m0 = blockIdx.y * 128;
    const int tid = threadIdx.x;
    const int wid = tid >> 5;
    const int lane = tid & 31;
    const int hl = lane >> 4;     // half-wave: 0/1
    const int ln = lane & 15;
    const int wm = wid & 3;       // wave M slot (x32)
    const int wn = wid >> 2;      // wave N slot (x64)

    v8f acc[2][4];
#pragma unroll
    for (int mt = 0; mt < 2; ++mt)
#pragma unroll
        for (int nt = 0; nt < 4; ++nt)
#pragma unroll
            for (int r = 0; r < 8; ++r) acc[mt][nt][r] = 0.f;

    const int arow = tid >> 1, acb = (tid & 1) * 16;   // A: 1 row, 32 cols per 2 threads
    const int brow = tid >> 3, bcb = (tid & 7) * 16;   // B: 1 row, 128 cols per 8 threads

    for (int k0 = 0; k0 < K; k0 += 32) {
        uint4 a0 = *(const uint4*)&A[(size_t)(m0 + arow) * K + k0 + acb];
        uint4 a1 = *(const uint4*)&A[(size_t)(m0 + arow) * K + k0 + acb + 8];
        uint4 b0 = *(const uint4*)&Bw[(size_t)(k0 + brow) * N + n0 + bcb];
        uint4 b1 = *(const uint4*)&Bw[(size_t)(k0 + brow) * N + n0 + bcb + 8];
        if (k0 + 32 < K) {  // hint next slab -> global_prefetch_b8
            __builtin_prefetch(&A[(size_t)(m0 + arow) * K + k0 + 32 + acb], 0, 1);
            __builtin_prefetch(&Bw[(size_t)(k0 + 32 + brow) * N + n0 + bcb], 0, 1);
        }
        __syncthreads();
        *(uint4*)&As[arow * 32 + acb] = a0;
        *(uint4*)&As[arow * 32 + acb + 8] = a1;
        {
            unsigned short tb[16];
            *(uint4*)tb = b0;
            *(uint4*)(tb + 8) = b1;
#pragma unroll
            for (int j = 0; j < 16; ++j) Bs[(bcb + j) * 32 + brow] = tb[j];
        }
        __syncthreads();

        v16bf af[2], bfv[4];
#pragma unroll
        for (int mt = 0; mt < 2; ++mt) {
            const int r = wm * 32 + mt * 16 + ln;
            af[mt] = frag_from(&As[r * 32 + hl * 8], &As[r * 32 + 16 + hl * 8]);
        }
#pragma unroll
        for (int nt = 0; nt < 4; ++nt) {
            const int c = wn * 64 + nt * 16 + ln;
            bfv[nt] = frag_from(&Bs[c * 32 + hl * 8], &Bs[c * 32 + 16 + hl * 8]);
        }
#pragma unroll
        for (int mt = 0; mt < 2; ++mt)
#pragma unroll
            for (int nt = 0; nt < 4; ++nt)
                acc[mt][nt] = wmma_bf16(af[mt], bfv[nt], acc[mt][nt]);
    }

#pragma unroll
    for (int mt = 0; mt < 2; ++mt)
#pragma unroll
        for (int nt = 0; nt < 4; ++nt)
#pragma unroll
            for (int r = 0; r < 8; ++r) {
                int row = m0 + wm * 32 + mt * 16 + r + hl * 8;
                int col = n0 + wn * 64 + nt * 16 + ln;
                size_t idx = (size_t)row * N + col;
                float v = acc[mt][nt][r];
                if (mode == 2)      Cf[idx] = v + bias[col] + resid[idx];
                else if (mode == 1) Cb[idx] = f2b(v);
                else                Cf[idx] = v;
            }
}

// ---------------- fused dual flash attention ----------------
// grid (SQ/128, HEADS, B); 256 threads = 8 waves; each wave owns 16 q rows.
__global__ void __launch_bounds__(256)
attn_kernel(const unsigned short* __restrict__ qb,
            const unsigned short* __restrict__ kbuf,   // [B*512, 768]
            const unsigned short* __restrict__ vbuf,   // [B*512, 768]
            const unsigned short* __restrict__ ipk,    // [B*64, 768]
            const unsigned short* __restrict__ ipv,    // [B*64, 768]
            unsigned short* __restrict__ attnb) {      // [B*2048, 1536]
    __shared__ unsigned short Ks[128 * 64];   // [key][dim]
    __shared__ unsigned short Vt[64 * 128];   // [dim][key] (transposed)
    __shared__ unsigned short Ps[8][16 * 32]; // per-wave P staging

    const int qt = blockIdx.x;
    const int h  = blockIdx.y;
    const int b  = blockIdx.z;
    const int kvh = h >> 1;                 // repeat_kv: 2 heads per kv head
    const int tid = threadIdx.x;
    const int wid = tid >> 5;
    const int lane = tid & 31;
    const int hl = lane >> 4;
    const int ln = lane & 15;

    // Q fragments (16 rows x 64 dims -> two 16x32 A frags), already scaled.
    const int qrow = qt * 128 + wid * 16 + ln;
    const unsigned short* qp =
        qb + ((size_t)(b * SQ + qrow)) * DMODEL + h * HD;
    v16bf qa[2];
#pragma unroll
    for (int c = 0; c < 2; ++c)
        qa[c] = frag_from(&qp[c * 32 + hl * 8], &qp[c * 32 + 16 + hl * 8]);

    float fo[4][8];
#pragma unroll
    for (int nt = 0; nt < 4; ++nt)
#pragma unroll
        for (int r = 0; r < 8; ++r) fo[nt][r] = 0.f;

    for (int pass = 0; pass < 2; ++pass) {
        const unsigned short* Kg = pass ? ipk : kbuf;
        const unsigned short* Vg = pass ? ipv : vbuf;
        const int bsr = pass ? IPL : TEXTL;  // rows per batch in K/V buffer
        const int nst = pass ? 1 : 4;        // stages of up to 128 keys
        const int rows = pass ? 64 : 128;

        float m[8], l[8];
        v8f o[4];
#pragma unroll
        for (int r = 0; r < 8; ++r) { m[r] = -3.0e38f; l[r] = 0.f; }
#pragma unroll
        for (int nt = 0; nt < 4; ++nt)
#pragma unroll
            for (int r = 0; r < 8; ++r) o[nt][r] = 0.f;

        for (int st = 0; st < nst; ++st) {
            __syncthreads();
            const int per = rows * 64 / 256;  // 32 (text) / 16 (ip) elems per thread
            // K stage: [key][dim] straight copy
            for (int e = tid * per, ee = 0; ee < per; ee += 8, e += 8) {
                int key = e >> 6, d = e & 63;
                uint4 v = *(const uint4*)&Kg[((size_t)(b * bsr + st * 128 + key)) * CROSS +
                                             kvh * HD + d];
                *(uint4*)&Ks[e] = v;
            }
            // V stage: transpose into [dim][key]
            for (int e = tid * per, ee = 0; ee < per; ee += 8, e += 8) {
                int key = e >> 6, d = e & 63;
                uint4 v = *(const uint4*)&Vg[((size_t)(b * bsr + st * 128 + key)) * CROSS +
                                             kvh * HD + d];
                unsigned short tmp[8];
                *(uint4*)tmp = v;
#pragma unroll
                for (int j = 0; j < 8; ++j) Vt[(d + j) * 128 + key] = tmp[j];
            }
            __syncthreads();

            const int nch = rows / 32;
            for (int cc = 0; cc < nch; ++cc) {
                const int kbase = cc * 32;
                // S = Q * K^T over 32 keys (two 16-key N tiles)
                v8f s[2];
#pragma unroll
                for (int t = 0; t < 2; ++t) {
#pragma unroll
                    for (int r = 0; r < 8; ++r) s[t][r] = 0.f;
#pragma unroll
                    for (int c = 0; c < 2; ++c) {
                        const unsigned short* kp =
                            &Ks[(kbase + t * 16 + ln) * 64 + c * 32 + hl * 8];
                        v16bf kf = frag_from(kp, kp + 16);
                        s[t] = wmma_bf16(qa[c], kf, s[t]);
                    }
                }
                // online softmax update (rows live across 16-lane halves)
#pragma unroll
                for (int r = 0; r < 8; ++r) {
                    float c0 = s[0][r], c1 = s[1][r];
                    float cm = redmax16(fmaxf(c0, c1));
                    float mn = fmaxf(m[r], cm);
                    float al = __expf(m[r] - mn);
                    float p0 = __expf(c0 - mn);
                    float p1 = __expf(c1 - mn);
                    float rsum = redsum16(p0 + p1);
                    l[r] = l[r] * al + rsum;
                    m[r] = mn;
#pragma unroll
                    for (int nt = 0; nt < 4; ++nt) o[nt][r] *= al;
                    const int prow = r + hl * 8;
                    Ps[wid][prow * 32 + ln] = f2b(p0);
                    Ps[wid][prow * 32 + 16 + ln] = f2b(p1);
                }
                // P (C/D layout) -> A layout via per-wave LDS staging
                v16bf pa = frag_from(&Ps[wid][ln * 32 + hl * 8],
                                     &Ps[wid][ln * 32 + 16 + hl * 8]);
                // O += P * V
#pragma unroll
                for (int nt = 0; nt < 4; ++nt) {
                    const unsigned short* vp =
                        &Vt[(nt * 16 + ln) * 128 + kbase + hl * 8];
                    v16bf vf = frag_from(vp, vp + 16);
                    o[nt] = wmma_bf16(pa, vf, o[nt]);
                }
            }
        }
        // finalize this attention and accumulate (IP_SCALE = 1)
#pragma unroll
        for (int r = 0; r < 8; ++r) {
            float inv = l[r] > 0.f ? 1.0f / l[r] : 0.f;
#pragma unroll
            for (int nt = 0; nt < 4; ++nt) fo[nt][r] += o[nt][r] * inv;
        }
    }

    // write [b, q, h*64 + d] as bf16 for the output projection
#pragma unroll
    for (int nt = 0; nt < 4; ++nt)
#pragma unroll
        for (int r = 0; r < 8; ++r) {
            int row = qt * 128 + wid * 16 + r + hl * 8;
            int dim = nt * 16 + ln;
            attnb[((size_t)(b * SQ + row)) * DMODEL + h * HD + dim] = f2b(fo[nt][r]);
        }
}

// ---------------- host side ----------------
static inline void* carve(char*& p, size_t bytes) {
    void* r = p;
    p += (bytes + 255) & ~(size_t)255;
    return r;
}

extern "C" void kernel_launch(void* const* d_in, const int* in_sizes, int n_in,
                              void* d_out, int out_size, void* d_ws, size_t ws_size,
                              hipStream_t stream) {
    (void)in_sizes; (void)n_in; (void)out_size; (void)ws_size;
    const float* hs   = (const float*)d_in[0];
    const float* enc  = (const float*)d_in[1];
    const float* rc   = (const float*)d_in[2];
    const float* rs   = (const float*)d_in[3];
    const float* wq   = (const float*)d_in[4];
    const float* wk   = (const float*)d_in[5];
    const float* wv   = (const float*)d_in[6];
    const float* wkip = (const float*)d_in[7];
    const float* wvip = (const float*)d_in[8];
    const float* wo   = (const float*)d_in[9];
    const float* bo   = (const float*)d_in[10];

    char* p = (char*)d_ws;
    const size_t NQ = (size_t)BATCH * SQ;          // 8192
    unsigned short* hsb   = (unsigned short*)carve(p, NQ * DMODEL * 2);
    unsigned short* wqb   = (unsigned short*)carve(p, (size_t)DMODEL * DMODEL * 2);
    unsigned short* wkb   = (unsigned short*)carve(p, (size_t)CROSS * CROSS * 2);
    unsigned short* wvb   = (unsigned short*)carve(p, (size_t)CROSS * CROSS * 2);
    unsigned short* wkipb = (unsigned short*)carve(p, (size_t)CROSS * CROSS * 2);
    unsigned short* wvipb = (unsigned short*)carve(p, (size_t)CROSS * CROSS * 2);
    unsigned short* wob   = (unsigned short*)carve(p, (size_t)DMODEL * DMODEL * 2);
    unsigned short* textb = (unsigned short*)carve(p, (size_t)BATCH * TEXTL * CROSS * 2);
    unsigned short* ipb   = (unsigned short*)carve(p, (size_t)BATCH * IPL * CROSS * 2);
    float*          qf    = (float*)carve(p, NQ * DMODEL * 4);
    unsigned short* qbuf  = (unsigned short*)carve(p, NQ * DMODEL * 2);
    unsigned short* kbuf  = (unsigned short*)carve(p, (size_t)BATCH * TEXTL * CROSS * 2);
    unsigned short* vbuf  = (unsigned short*)carve(p, (size_t)BATCH * TEXTL * CROSS * 2);
    unsigned short* ipkb  = (unsigned short*)carve(p, (size_t)BATCH * IPL * CROSS * 2);
    unsigned short* ipvb  = (unsigned short*)carve(p, (size_t)BATCH * IPL * CROSS * 2);
    unsigned short* attnb = (unsigned short*)carve(p, NQ * DMODEL * 2);

    const int T = 256;
    auto blocks = [](size_t n) { return (unsigned)((n + 255) / 256); };

    // 1) pack activations / weights to bf16
    cvt_bf16_kernel<<<blocks(NQ * DMODEL), T, 0, stream>>>(hs, hsb, (int)(NQ * DMODEL));
    cvt_bf16_kernel<<<blocks((size_t)DMODEL * DMODEL), T, 0, stream>>>(wq, wqb, DMODEL * DMODEL);
    cvt_bf16_kernel<<<blocks((size_t)CROSS * CROSS), T, 0, stream>>>(wk, wkb, CROSS * CROSS);
    cvt_bf16_kernel<<<blocks((size_t)CROSS * CROSS), T, 0, stream>>>(wv, wvb, CROSS * CROSS);
    cvt_bf16_kernel<<<blocks((size_t)CROSS * CROSS), T, 0, stream>>>(wkip, wkipb, CROSS * CROSS);
    cvt_bf16_kernel<<<blocks((size_t)CROSS * CROSS), T, 0, stream>>>(wvip, wvipb, CROSS * CROSS);
    cvt_bf16_kernel<<<blocks((size_t)DMODEL * DMODEL), T, 0, stream>>>(wo, wob, DMODEL * DMODEL);
    enc_split_kernel<<<blocks((size_t)BATCH * ENCL * CROSS), T, 0, stream>>>(enc, textb, ipb);

    // 2) Q = hs @ wq (f32 out for RoPE)
    gemm_bf16_wmma<<<dim3(DMODEL / 128, NQ / 128), T, 0, stream>>>(
        hsb, wqb, qf, nullptr, nullptr, nullptr, (int)NQ, DMODEL, DMODEL, 0);
    // 3) RoPE + scale + pack
    rope_pack_kernel<<<blocks(NQ * DMODEL), T, 0, stream>>>(qf, rc, rs, qbuf);

    // 4) K/V projections (bf16 out)
    gemm_bf16_wmma<<<dim3(CROSS / 128, BATCH * TEXTL / 128), T, 0, stream>>>(
        textb, wkb, nullptr, kbuf, nullptr, nullptr, BATCH * TEXTL, CROSS, CROSS, 1);
    gemm_bf16_wmma<<<dim3(CROSS / 128, BATCH * TEXTL / 128), T, 0, stream>>>(
        textb, wvb, nullptr, vbuf, nullptr, nullptr, BATCH * TEXTL, CROSS, CROSS, 1);
    gemm_bf16_wmma<<<dim3(CROSS / 128, BATCH * IPL / 128), T, 0, stream>>>(
        ipb, wkipb, nullptr, ipkb, nullptr, nullptr, BATCH * IPL, CROSS, CROSS, 1);
    gemm_bf16_wmma<<<dim3(CROSS / 128, BATCH * IPL / 128), T, 0, stream>>>(
        ipb, wvipb, nullptr, ipvb, nullptr, nullptr, BATCH * IPL, CROSS, CROSS, 1);

    // 5) fused text + IP attention
    attn_kernel<<<dim3(SQ / 128, HEADS, BATCH), T, 0, stream>>>(
        qbuf, kbuf, vbuf, ipkb, ipvb, attnb);

    // 6) out = attn @ wo + bo + residual
    gemm_bf16_wmma<<<dim3(DMODEL / 128, NQ / 128), T, 0, stream>>>(
        attnb, wob, (float*)d_out, nullptr, bo, hs, (int)NQ, DMODEL, DMODEL, 2);
}
